// net_3152505995579
// MI455X (gfx1250) — matrix operand order repaired and checked by reading, not compiled, not run on hardware
//
#include <hip/hip_runtime.h>

// ---------------------------------------------------------------------------
// MI455X (gfx1250) CNN pipeline, all matmul stages via v_wmma_f32_16x16x32_bf16
//
//   conv1: [B,3,16,16] -> [B,10,14,14] relu   (implicit GEMM, K=27, N=10)
//   conv2: -> [B,20,12,12] relu               (implicit GEMM, K=90, N=20)
//   fc1  : [B,2880] x [2880,500] + b, step    (tiled WMMA GEMM, N pad 512,
//                                              B-tile staged via TDM tensor_load_to_lds)
//   fc2  : [B,500] x [500,10] + b             (WMMA GEMM, K pad 512, N pad 16)
//
// Workspace: ~426.8 MiB required in d_ws.
// ---------------------------------------------------------------------------

typedef __attribute__((ext_vector_type(16))) __bf16 v16bf;
typedef __attribute__((ext_vector_type(8)))  __bf16 v8bf;
typedef __attribute__((ext_vector_type(8)))  float  v8f;
typedef unsigned int u32x4 __attribute__((ext_vector_type(4)));
typedef int          i32x4 __attribute__((ext_vector_type(4)));
typedef int          i32x8 __attribute__((ext_vector_type(8)));

#define WMMA_BF16(a, b, c) \
  __builtin_amdgcn_wmma_f32_16x16x32_bf16(false, (a), false, (b), (short)0, (c), false, false)

// Tensor Data Mover availability (arity differs across toolchains; see bridge doc)
#if defined(__has_builtin)
#if __has_builtin(__builtin_amdgcn_tensor_load_to_lds) && \
    __has_builtin(__builtin_amdgcn_s_wait_tensorcnt)
#define CDNA5_HAS_TDM 1
#endif
#endif
#ifndef CDNA5_HAS_TDM
#define CDNA5_HAS_TDM 0
#endif

// Workspace offsets in bf16 elements (all 32B aligned)
static constexpr long OFF_W1  = 0;                         // [32 lanes][16]            = 512
static constexpr long OFF_W2  = 512;                       // [3 kc][2 nt][32][16]      = 3072
static constexpr long OFF_FC2 = 3584;                      // [16 kc][32][16]           = 8192
static constexpr long OFF_WT  = 11776;                     // [90 kc][32 nt][32][16]    = 1474560
static constexpr long OFF_H2  = 1486336;                   // [65536][2880]             = 188743680
static constexpr long OFF_S   = 190230016;                 // [65536][512]              = 33554432
static constexpr long N_PREP  = 1486336;
static constexpr int  BATCH   = 65536;

// ---------------------------------------------------------------------------
// Kernel 1: convert all weights to bf16, pre-swizzled into WMMA B-fragment
// layout: element i of lane L holds B[K = (i&7) + 8*(L>>4) + 16*(i>>3)][N = (L&15)].
// ---------------------------------------------------------------------------
__global__ __launch_bounds__(256) void prep_weights(
    const float* __restrict__ w1, const float* __restrict__ w2,
    const float* __restrict__ fc_w, const float* __restrict__ fc2_w,
    __bf16* __restrict__ wsb)
{
  for (long idx = (long)blockIdx.x * 256 + threadIdx.x; idx < N_PREP;
       idx += (long)gridDim.x * 256) {
    float v = 0.f;
    long o;
    if (idx < 512) {                                   // conv1 B-frag: K=27->32, N=10->16
      int lane = (int)(idx >> 4), i = (int)(idx & 15);
      int k = 8 * (lane >> 4) + (i & 7) + 16 * (i >> 3), n = lane & 15;
      if (n < 10 && k < 27) v = w1[n * 27 + k];
      o = OFF_W1 + idx;
    } else if (idx < 3584) {                           // conv2 B-frags: K=90->96, N=20->32
      long t = idx - 512;
      int i = (int)(t & 15), lane = (int)((t >> 4) & 31);
      int nt = (int)((t >> 9) & 1), kc = (int)(t >> 10);
      int k = kc * 32 + 8 * (lane >> 4) + (i & 7) + 16 * (i >> 3), n = nt * 16 + (lane & 15);
      if (n < 20 && k < 90) v = w2[n * 90 + k];
      o = OFF_W2 + t;
    } else if (idx < 11776) {                          // fc2 B-frags: K=500->512, N=10->16
      long t = idx - 3584;
      int i = (int)(t & 15), lane = (int)((t >> 4) & 31), kc = (int)(t >> 9);
      int k = kc * 32 + 8 * (lane >> 4) + (i & 7) + 16 * (i >> 3), n = lane & 15;
      if (n < 10 && k < 500) v = fc2_w[n * 500 + k];
      o = OFF_FC2 + t;
    } else {                                           // fc1 B-frags: K=2880, N=500->512
      long t = idx - 11776;
      int i = (int)(t & 15), lane = (int)((t >> 4) & 31);
      int nt = (int)((t >> 9) & 31), kc = (int)(t >> 14);
      int k = kc * 32 + 8 * (lane >> 4) + (i & 7) + 16 * (i >> 3), n = nt * 16 + (lane & 15);
      if (n < 500) v = fc_w[(long)n * 2880 + k];
      o = OFF_WT + t;
    }
    wsb[o] = (__bf16)v;
  }
}

// ---------------------------------------------------------------------------
// Kernel 2: fused conv1+relu+conv2+relu per image. 2 waves / image.
// LDS arrays are padded so all epilogue stores are unconditional 16B
// ds_store_b128 (no EXEC-juggling); pad rows/channels are never read.
// ---------------------------------------------------------------------------
__global__ __launch_bounds__(64) void conv_fused(
    const float* __restrict__ x, const float* __restrict__ b1,
    const float* __restrict__ b2, const __bf16* __restrict__ wsb,
    __bf16* __restrict__ h2out)
{
  __shared__ __bf16 xs[3 * 256];                                   // input image, bf16
  __shared__ __attribute__((aligned(16))) __bf16 h1[16 * 208];     // conv1 out, padded
  __shared__ __attribute__((aligned(16))) __bf16 h2[32 * 144];     // conv2 out, padded

  const int tid  = threadIdx.x;
  const int lane = tid & 31;
  const int wave = tid >> 5;
  const int half = lane >> 4;
  const int ln   = lane & 15;
  const long b   = blockIdx.x;

  const float* xp = x + b * 768;
  for (int i = tid; i < 768; i += 64) xs[i] = (__bf16)xp[i];
  __syncthreads();

  // ---- conv1: M = 196 positions (13 tiles of 16), K = 27, N = 10 ----
  v16bf w1f = *(const v16bf*)(wsb + OFF_W1 + lane * 16);
  float bias1 = (ln < 10) ? b1[ln] : 0.f;

  for (int mt = wave; mt < 13; mt += 2) {
    int p  = mt * 16 + ln;
    int pc = (p < 196) ? p : 0;                        // clamp pad rows (discarded)
    int oy = pc / 14, ox = pc % 14;
    union { v16bf v; __bf16 e[16]; } a;
#pragma unroll
    for (int i = 0; i < 16; ++i) {
      int k  = (i & 7) + 16 * (i >> 3) + 8 * half;     // A-layout K index
      int kk = (k < 27) ? k : 0;                       // pad K multiplies zero weight
      int c = kk / 9, rr = (kk % 9) / 3, ss = kk % 3;
      a.e[i] = xs[c * 256 + (oy + rr) * 16 + (ox + ss)];
    }
    v8f acc = {};
    acc = WMMA_BF16(a.v, w1f, acc);
    union { v8bf v; __bf16 e[8]; } o;
#pragma unroll
    for (int r = 0; r < 8; ++r) {
      float vv = acc[r] + bias1;
      o.e[r] = (__bf16)(vv > 0.f ? vv : 0.f);
    }
    // rows r..r+7 are consecutive positions -> one aligned 16B LDS store
    *(v8bf*)(h1 + ln * 208 + mt * 16 + 8 * half) = o.v;
  }
  __syncthreads();

  // ---- conv2: M = 144 positions (9 tiles), K = 90 (3 chunks), N = 20 (2 tiles) ----
  v16bf w2f[3][2];
#pragma unroll
  for (int kc = 0; kc < 3; ++kc)
#pragma unroll
    for (int nt = 0; nt < 2; ++nt)
      w2f[kc][nt] = *(const v16bf*)(wsb + OFF_W2 + ((kc * 2 + nt) * 32 + lane) * 16);
  float bias2a = b2[ln];                               // channels 0..15
  float bias2b = (16 + ln < 20) ? b2[16 + ln] : 0.f;   // channels 16..19

  for (int mt = wave; mt < 9; mt += 2) {
    int p = mt * 16 + ln;
    int oy = p / 12, ox = p % 12;
    v8f acc0 = {}, acc1 = {};
#pragma unroll
    for (int kc = 0; kc < 3; ++kc) {
      union { v16bf v; __bf16 e[16]; } a;
#pragma unroll
      for (int i = 0; i < 16; ++i) {
        int k  = kc * 32 + (i & 7) + 16 * (i >> 3) + 8 * half;
        int kk = (k < 90) ? k : 0;
        int c = kk / 9, rr = (kk % 9) / 3, ss = kk % 3;
        a.e[i] = h1[c * 208 + (oy + rr) * 14 + (ox + ss)];
      }
      acc0 = WMMA_BF16(a.v, w2f[kc][0], acc0);
      acc1 = WMMA_BF16(a.v, w2f[kc][1], acc1);
    }
    union { v8bf v; __bf16 e[8]; } o0, o1;
#pragma unroll
    for (int r = 0; r < 8; ++r) {
      float v0 = acc0[r] + bias2a;
      float v1 = acc1[r] + bias2b;
      o0.e[r] = (__bf16)(v0 > 0.f ? v0 : 0.f);
      o1.e[r] = (__bf16)(v1 > 0.f ? v1 : 0.f);
    }
    *(v8bf*)(h2 + ln * 144 + mt * 16 + 8 * half)        = o0.v;  // ch 0..15
    *(v8bf*)(h2 + (16 + ln) * 144 + mt * 16 + 8 * half) = o1.v;  // ch 16..31 (>=20 pad)
  }
  __syncthreads();

  // coalesced writeout of the NCHW-flattened [2880] bf16 row as dwords
  const unsigned int* src = (const unsigned int*)h2;
  unsigned int* dst = (unsigned int*)(h2out + b * 2880);
  for (int i = tid; i < 1440; i += 64) dst[i] = src[i];
}

// ---------------------------------------------------------------------------
// Kernel 3: FC1 GEMM [B,2880] x [2880,512] + bias, fused Heaviside -> bf16 {0,1}.
// Block tile 128(M) x 64(N); 8 waves x (32x32) sub-tiles. A staged to LDS in
// fragment layout; B-tile (4 KB contiguous pre-swizzled fragments) staged by
// the Tensor Data Mover when available, with global prefetch of the next A K-chunk.
// ---------------------------------------------------------------------------
__global__ __launch_bounds__(256) void fc1_step(
    const float* __restrict__ fc_b, const __bf16* __restrict__ wsb,
    __bf16* __restrict__ sout)
{
  __shared__ __attribute__((aligned(32))) __bf16 At[8 * 32 * 16];  // 8 M-tiles
  __shared__ __attribute__((aligned(32))) __bf16 Bt[4 * 32 * 16];  // 4 N-tiles

  const int tid  = threadIdx.x;
  const int lane = tid & 31, wave = tid >> 5;
  const int half = lane >> 4, ln = lane & 15;
  const int mw = wave & 3, nw = wave >> 2;             // 4x2 wave grid
  const int mBlock = blockIdx.x >> 3;                  // 512 M-blocks of 128 rows
  const int nBlock = blockIdx.x & 7;                   // 8 N-blocks of 64 cols
  const __bf16* h2 = wsb + OFF_H2;
  const __bf16* wt = wsb + OFF_WT;

  v8f acc00 = {}, acc01 = {}, acc10 = {}, acc11 = {};

  for (int kc = 0; kc < 90; ++kc) {
    // ---- B tile: 4 KB contiguous fragment block ----
    const __bf16* bsrc = wt + ((long)kc * 32 + nBlock * 4) * 32 * 16;
#if CDNA5_HAS_TDM
    if (wave == 0) {
      // Tensor DMA descriptor (CDNA5 ISA):
      // 1-D tile: data_size=8B, tile_dim0=512 units (=4096B), single line.
      unsigned lds_off = (unsigned)(unsigned long long)(const void*)&Bt[0];
      unsigned long long ga = (unsigned long long)(const void*)bsrc;
      u32x4 g0;
      g0[0] = 1u;                                        // count=1 (valid), user mode
      g0[1] = lds_off;                                   // lds_addr
      g0[2] = (unsigned)ga;                              // global_addr[31:0]
      g0[3] = (unsigned)((ga >> 32) & 0x01FFFFFFull) | 0x80000000u; // ga[56:32]|type=2
      i32x8 g1;
      g1[0] = 0x00030000;                                // data_size=3 (8B), mask=0
      g1[1] = (int)(512u << 16);                         // tensor_dim0 = 512
      g1[2] = 0x00010000;                                // tensor_dim1 = 1
      g1[3] = 0x02000000;                                // tile_dim0 = 512
      g1[4] = 0x00010001;                                // tile_dim1 = 1, tile_dim2 = 1
      g1[5] = 512;                                       // tensor_dim0_stride = 512
      g1[6] = 0x02000000;                                // tensor_dim1_stride = 512
      g1[7] = 0;
      i32x4 g2 = {0, 0, 0, 0}, g3 = {0, 0, 0, 0};
#if __clang_major__ >= 23
      i32x8 g4 = {0, 0, 0, 0, 0, 0, 0, 0};
      __builtin_amdgcn_tensor_load_to_lds(g0, g1, g2, g3, g4, 0);
#else
      __builtin_amdgcn_tensor_load_to_lds(g0, g1, g2, g3, 0);
#endif
    }
#else
    {
      int nt = tid >> 6, rest = tid & 63;
      int l = rest >> 1, i0 = (rest & 1) * 8;
      const uint4* src = (const uint4*)(bsrc + (nt * 32 + l) * 16 + i0);
      *(uint4*)(Bt + (nt * 32 + l) * 16 + i0) = *src;
    }
#endif

    // ---- A tile: 128 rows x 32 K, placed directly in fragment layout ----
    // group (m, kg): lane=(m&15)+16*(kg&1), i0=(kg>>1)*8, K off=(kg>>1)*16+(kg&1)*8
#pragma unroll
    for (int u = 0; u < 2; ++u) {
      int idx = tid + u * 256;
      int m = idx >> 2, kg = idx & 3;
      int l = (m & 15) + 16 * (kg & 1);
      int i0 = (kg >> 1) * 8;
      long row = (long)mBlock * 128 + m;
      const __bf16* asrc = h2 + row * 2880 + kc * 32 + (kg >> 1) * 16 + (kg & 1) * 8;
      *(uint4*)(At + ((m >> 4) * 32 + l) * 16 + i0) = *(const uint4*)asrc;
      if (kc + 1 < 90) __builtin_prefetch(asrc + 32, 0, 1);  // next K-chunk, same row
    }

#if CDNA5_HAS_TDM
    if (wave == 0) __builtin_amdgcn_s_wait_tensorcnt(0);     // TDM done before barrier
#endif
    __syncthreads();

    v16bf a0 = *(const v16bf*)(At + ((2 * mw + 0) * 32 + lane) * 16);
    v16bf a1 = *(const v16bf*)(At + ((2 * mw + 1) * 32 + lane) * 16);
    v16bf b0 = *(const v16bf*)(Bt + ((2 * nw + 0) * 32 + lane) * 16);
    v16bf b1 = *(const v16bf*)(Bt + ((2 * nw + 1) * 32 + lane) * 16);
    acc00 = WMMA_BF16(a0, b0, acc00);
    acc01 = WMMA_BF16(a0, b1, acc01);
    acc10 = WMMA_BF16(a1, b0, acc10);
    acc11 = WMMA_BF16(a1, b1, acc11);
    __syncthreads();
  }

  // epilogue: bias + Heaviside step; pad columns (>=500) forced to 0
  int n0 = nBlock * 64 + (2 * nw) * 16 + ln;
  int n1 = n0 + 16;
  float bb0 = (n0 < 500) ? fc_b[n0] : 0.f;
  float bb1 = (n1 < 500) ? fc_b[n1] : 0.f;
  long mbase = (long)mBlock * 128;
#pragma unroll
  for (int r = 0; r < 8; ++r) {
    long r0 = mbase + (2 * mw + 0) * 16 + r + 8 * half;
    long r1 = mbase + (2 * mw + 1) * 16 + r + 8 * half;
    sout[r0 * 512 + n0] = (__bf16)((n0 < 500 && acc00[r] + bb0 >= 0.f) ? 1.f : 0.f);
    sout[r0 * 512 + n1] = (__bf16)((n1 < 500 && acc01[r] + bb1 >= 0.f) ? 1.f : 0.f);
    sout[r1 * 512 + n0] = (__bf16)((n0 < 500 && acc10[r] + bb0 >= 0.f) ? 1.f : 0.f);
    sout[r1 * 512 + n1] = (__bf16)((n1 < 500 && acc11[r] + bb1 >= 0.f) ? 1.f : 0.f);
  }
}

// ---------------------------------------------------------------------------
// Kernel 4: FC2 GEMM [B,512] x [512,16] + bias -> f32 out [B,10].
// One wave per 16-row tile; A fragments loaded straight from global.
// ---------------------------------------------------------------------------
__global__ __launch_bounds__(256) void fc2_kernel(
    const float* __restrict__ fc2_b, const __bf16* __restrict__ wsb,
    float* __restrict__ out)
{
  const int tid  = threadIdx.x;
  const int lane = tid & 31, wave = tid >> 5;
  const int half = lane >> 4, ln = lane & 15;
  long mt   = (long)blockIdx.x * 8 + wave;             // 4096 M-tiles total
  long row0 = mt * 16;
  const __bf16* s = wsb + OFF_S;
  const __bf16* w = wsb + OFF_FC2;

  v8f acc = {};
#pragma unroll 4
  for (int kc = 0; kc < 16; ++kc) {
    union { v16bf v; v8bf h[2]; } a;
    long base = (row0 + ln) * 512 + kc * 32 + 8 * half;
    a.h[0] = *(const v8bf*)(s + base);                 // elements 0..7  : K = 8*half+0..7
    a.h[1] = *(const v8bf*)(s + base + 16);            // elements 8..15 : K = 16+8*half+0..7
    v16bf bf = *(const v16bf*)(w + (kc * 32 + lane) * 16);
    acc = WMMA_BF16(a.v, bf, acc);
  }
  float bias = (ln < 10) ? fc2_b[ln] : 0.f;
#pragma unroll
  for (int r = 0; r < 8; ++r) {
    long rr = row0 + r + 8 * half;
    if (ln < 10) out[rr * 10 + ln] = acc[r] + bias;
  }
}

// ---------------------------------------------------------------------------
// Launch. Requires ws_size >= ~447.6 MB.
// ---------------------------------------------------------------------------
extern "C" void kernel_launch(void* const* d_in, const int* in_sizes, int n_in,
                              void* d_out, int out_size, void* d_ws, size_t ws_size,
                              hipStream_t stream) {
  const float* x     = (const float*)d_in[0];
  const float* w1    = (const float*)d_in[1];
  const float* b1    = (const float*)d_in[2];
  const float* w2    = (const float*)d_in[3];
  const float* b2    = (const float*)d_in[4];
  const float* fc_w  = (const float*)d_in[5];
  const float* fc_b  = (const float*)d_in[6];
  const float* fc2_w = (const float*)d_in[7];
  const float* fc2_b = (const float*)d_in[8];
  float* out  = (float*)d_out;
  __bf16* wsb = (__bf16*)d_ws;

  prep_weights<<<2048, 256, 0, stream>>>(w1, w2, fc_w, fc2_w, wsb);
  conv_fused<<<BATCH, 64, 0, stream>>>(x, b1, b2, wsb, wsb + OFF_H2);
  fc1_step<<<4096, 256, 0, stream>>>(fc_b, wsb, wsb + OFF_S);
  fc2_kernel<<<512, 256, 0, stream>>>(fc2_b, wsb, out);
}
